// CoOccurrenceGraph_84791244357655
// MI455X (gfx1250) — matrix operand order repaired and checked by reading, not compiled, not run on hardware
//
#include <hip/hip_runtime.h>

// ---------------------------------------------------------------------------
// CoOccurrenceGraph for MI455X (gfx1250, wave32, WMMA).
//
// Roofline: GEMM = 2*64*512*512*1024 = 34.4 GFLOP; min HBM = ~270 MB -> ~12us.
// bf16 WMMA (16x16x32, 8192 MAC/inst) has 8x the density of f32 WMMA
// (16x16x4); x[b] slices (2MB) re-read per row-tile stay in the 192MB L2.
// GEMM runs in bf16 with f32 accumulation; gate epilogue is fused so x_t
// (134 MB) never touches HBM.
//
// Round-1 fixes: M-tile 16 -> 32 (halves staging work & LDS traffic per
// WMMA), exact v_cvt_pk_bf16_f32 pairing via inline asm (kills the
// v_mov_b16/v_or fixup chains the compiler generated), global_prefetch of
// the next K-step's rows.
// ---------------------------------------------------------------------------

typedef __attribute__((ext_vector_type(16))) __bf16 v16bf;
typedef __attribute__((ext_vector_type(8)))  float  v8f;

#define NCLS 512
#define FEAT 1024
#define NBATCH 64
#define SM 0.01f

__device__ __forceinline__ unsigned short f2bf(float f) {
  union { __bf16 h; unsigned short u; } v;
  v.h = (__bf16)f;
  return v.u;
}

// Exact packed convert: lo -> [15:0], hi -> [31:16]. One VALU op.
__device__ __forceinline__ unsigned int pack2bf(float lo, float hi) {
  unsigned int r;
  asm("v_cvt_pk_bf16_f32 %0, %1, %2" : "=v"(r) : "v"(lo), "v"(hi));
  return r;
}

__device__ __forceinline__ float sigmoidf_(float z) {
  return 1.0f / (1.0f + __expf(-z));
}

// ---------------------------------------------------------------------------
// Kernel 1: build edge weights (512x512), store as bf16 into workspace.
// One block per row i; trivially cheap vs the GEMM.
// ---------------------------------------------------------------------------
__global__ __launch_bounds__(256) void build_edge_weights(
    const float* __restrict__ co, const float* __restrict__ counts,
    const float* __restrict__ emb, unsigned short* __restrict__ ew) {
  __shared__ float red[256];
  __shared__ float srow[NCLS];
  const int i = blockIdx.x;
  const int tid = threadIdx.x;

  // mean(counts)
  red[tid] = counts[tid] + counts[tid + 256];
  __syncthreads();
  for (int s = 128; s > 0; s >>= 1) {
    if (tid < s) red[tid] += red[tid + s];
    __syncthreads();
  }
  const float avg = red[0] * (1.0f / 512.0f);
  __syncthreads();

  const float ci = counts[i];
  const float4 ei = *(const float4*)(emb + i * 4);
  const float inv_ni =
      rsqrtf(ei.x * ei.x + ei.y * ei.y + ei.z * ei.z + ei.w * ei.w);

  for (int j = tid; j < NCLS; j += 256) {
    float s = 0.0f;
    if (j != i) {
      const float cj = counts[j];
      const float coij = co[i * NCLS + j];
      const float nco = (coij + SM) * rsqrtf((ci + SM) * (cj + SM));
      const float4 ej = *(const float4*)(emb + j * 4);
      const float inv_nj =
          rsqrtf(ej.x * ej.x + ej.y * ej.y + ej.z * ej.z + ej.w * ej.w);
      const float sim =
          (ei.x * ej.x + ei.y * ej.y + ei.z * ej.z + ei.w * ej.w) * inv_ni *
          inv_nj;
      const float aff = sim * sigmoidf_((sim - 0.5f) * 10.0f);
      const float mn = fminf(ci, cj), mx = fmaxf(ci, cj);
      const float bal =
          ((mn > SM) && (mx > SM)) ? log1pf(mx / avg) * (mn / mx) : SM;
      const float conf = 2.0f * sigmoidf_(coij * (1.0f / 5.0f)) - 1.0f;
      s = nco * aff * bal * conf;
    }
    srow[j] = s * 5.0f;  // softmax(ew * 5)
  }
  __syncthreads();

  // row softmax
  red[tid] = fmaxf(srow[tid], srow[tid + 256]);
  __syncthreads();
  for (int s = 128; s > 0; s >>= 1) {
    if (tid < s) red[tid] = fmaxf(red[tid], red[tid + s]);
    __syncthreads();
  }
  const float rmax = red[0];
  __syncthreads();
  red[tid] = __expf(srow[tid] - rmax) + __expf(srow[tid + 256] - rmax);
  __syncthreads();
  for (int s = 128; s > 0; s >>= 1) {
    if (tid < s) red[tid] += red[tid + s];
    __syncthreads();
  }
  const float inv_sum = 1.0f / red[0];
  for (int j = tid; j < NCLS; j += 256) {
    const float p =
        __expf(srow[j] - rmax) * inv_sum * 0.9f + ((j == i) ? 0.1f : 0.0f);
    ew[i * NCLS + j] = f2bf(p);
  }
}

// ---------------------------------------------------------------------------
// Kernel 2: fused  x_t = ew @ x[b]  (bf16 WMMA, f32 accum)  + gate + blend.
//
// Block = (batch b, 32-row tile i0). 512 threads = 16 waves:
//   mhalf = wave>>3 selects the 16-row half, wcol = wave&7 selects a
//   64-column group within each 512-wide d-phase.
// Each wave holds 8 f32 accumulator tiles (64 VGPRs); per K-step of 32 it
// issues 8 WMMAs against one A fragment and 8 LDS B fragments.
//
// LDS holds bf16 (x[j], x[j+1]) K-pairs pre-packed in the exact WMMA B
// layout:  word(d, jp) = (d>>2)*84 + (d&3)*20 + jp  (jp = pair index 0..15)
// -> fragment reads are 16B-aligned ds_load_b128 pairs; pack-stores are at
//    worst 2-way bank conflicted.
// ---------------------------------------------------------------------------
#define GRP 84   // words per 4-column group (4*20 used + 4 pad)
#define ROWW 20  // words per column within a group (16 pairs + 4 pad)

__device__ __forceinline__ int lds_word(int d, int jp) {
  return (d >> 2) * GRP + (d & 3) * ROWW + jp;
}

__global__ __launch_bounds__(512) void fused_graph_gate(
    const float* __restrict__ x, const unsigned short* __restrict__ ew,
    float* __restrict__ out) {
  __shared__ __align__(16) unsigned int xs[128 * GRP];  // 43008 B
  __shared__ float gdot[32];

  const int tid = threadIdx.x;
  const int wave = tid >> 5;
  const int lane = tid & 31;
  const int n = lane & 15;      // N column / A row within tile
  const int khalf = lane >> 4;  // lane half selects K sub-range
  const int mhalf = wave >> 3;  // which 16-row half of the 32-row tile
  const int wcol = wave & 7;    // which 64-column group

  const int b = blockIdx.x >> 4;          // 64 batches
  const int i0 = (blockIdx.x & 15) << 5;  // 16 row tiles of 32
  const float* __restrict__ xb = x + (size_t)b * NCLS * FEAT;

  if (tid < 32) gdot[tid] = 0.0f;

  v8f acc[8] = {};  // 8 N-tiles x 8 VGPRs of f32 accumulators

  const int jp = tid >> 5;     // staging: pair-of-rows index 0..15
  const int dlane = tid & 31;  // staging: column sub-lane

  for (int k0 = 0; k0 < NCLS; k0 += 32) {
    // --- A fragment: ew[i0 + mhalf*16 + n][k0..k0+31], in this code's contiguous
    // K order (k = 16*khalf + s), same as the B fragments:
    //   lanes 0-15 : V0-3 = K 0..7,   V4-7 = K 8..15
    //   lanes 16-31: V0-3 = K 16..23, V4-7 = K 24..31
    union { uint4 u[2]; v16bf v; } afrag;
    const unsigned short* arow =
        ew + (size_t)(i0 + mhalf * 16 + n) * NCLS + k0 + khalf * 16;
    afrag.u[0] = *(const uint4*)(arow);
    afrag.u[1] = *(const uint4*)(arow + 8);

    // pull next K-step's staging rows toward the WGP while WMMAs run
    if (k0 + 32 < NCLS) {
      __builtin_prefetch(xb + (size_t)(k0 + 32 + 2 * jp) * FEAT + dlane * 16,
                         0, 1);
    }

    #pragma unroll
    for (int phase = 0; phase < 2; ++phase) {
      __syncthreads();  // previous phase's fragment reads complete
      // Stage x[b][k0..k0+31][phase*512 .. +511] as packed bf16 K-pairs.
      // One wave per row-pair; float4 global loads are 512B-coalesced.
      const float* r0 = xb + (size_t)(k0 + 2 * jp) * FEAT + phase * 512;
      const float* r1 = r0 + FEAT;
      #pragma unroll
      for (int it = 0; it < 4; ++it) {
        const int d = dlane * 4 + it * 128;
        const float4 a0 = *(const float4*)(r0 + d);
        const float4 a1 = *(const float4*)(r1 + d);
        xs[lds_word(d + 0, jp)] = pack2bf(a0.x, a1.x);
        xs[lds_word(d + 1, jp)] = pack2bf(a0.y, a1.y);
        xs[lds_word(d + 2, jp)] = pack2bf(a0.z, a1.z);
        xs[lds_word(d + 3, jp)] = pack2bf(a0.w, a1.w);
      }
      __syncthreads();
      // --- 4 WMMAs for this phase's N-tiles
      #pragma unroll
      for (int t = 0; t < 4; ++t) {
        const int col = wcol * 64 + t * 16 + n;  // column within phase
        const int base = (col >> 2) * GRP + (col & 3) * ROWW + khalf * 8;
        union { uint4 u[2]; v16bf v; } bfrag;
        bfrag.u[0] = *(const uint4*)(&xs[base]);
        bfrag.u[1] = *(const uint4*)(&xs[base + 4]);
        const int tile = phase * 4 + t;
        acc[tile] = __builtin_amdgcn_wmma_f32_16x16x32_bf16(
            false, afrag.v, false, bfrag.v, (short)0, acc[tile], false, false);
      }
    }
  }

  // --- Epilogue: gate = sigmoid(sum_d x*x_t / 32); out = lerp(x, x_t, gate)
  // C/D layout: VGPR r, lanes 0-15 -> row r; lanes 16-31 -> row r+8.
  float xv[8][8];
  float partial[8];
  #pragma unroll
  for (int r = 0; r < 8; ++r) partial[r] = 0.0f;

  const int rowbase = i0 + mhalf * 16 + khalf * 8;
  #pragma unroll
  for (int tile = 0; tile < 8; ++tile) {
    const int col = (tile >> 2) * 512 + wcol * 64 + (tile & 3) * 16 + n;
    #pragma unroll
    for (int r = 0; r < 8; ++r) {
      const float xvv = xb[(size_t)(rowbase + r) * FEAT + col];
      xv[tile][r] = xvv;
      partial[r] += xvv * acc[tile][r];
    }
  }
  // reduce over the 16 N lanes (stays within each 16-lane half of the wave32)
  #pragma unroll
  for (int r = 0; r < 8; ++r) {
    float p = partial[r];
    p += __shfl_xor(p, 1, 32);
    p += __shfl_xor(p, 2, 32);
    p += __shfl_xor(p, 4, 32);
    p += __shfl_xor(p, 8, 32);
    partial[r] = p;
  }
  if (n == 0) {
    #pragma unroll
    for (int r = 0; r < 8; ++r)
      atomicAdd(&gdot[mhalf * 16 + khalf * 8 + r], partial[r]);
  }
  __syncthreads();

  float* __restrict__ outb = out + (size_t)b * NCLS * FEAT;
  #pragma unroll
  for (int r = 0; r < 8; ++r) {
    const float g =
        sigmoidf_(gdot[mhalf * 16 + khalf * 8 + r] * (1.0f / 32.0f));
    #pragma unroll
    for (int tile = 0; tile < 8; ++tile) {
      const int col = (tile >> 2) * 512 + wcol * 64 + (tile & 3) * 16 + n;
      outb[(size_t)(rowbase + r) * FEAT + col] =
          xv[tile][r] * (1.0f - g) + acc[tile][r] * g;
    }
  }
}

// ---------------------------------------------------------------------------
extern "C" void kernel_launch(void* const* d_in, const int* in_sizes, int n_in,
                              void* d_out, int out_size, void* d_ws,
                              size_t ws_size, hipStream_t stream) {
  const float* x      = (const float*)d_in[0];  // [64, 512, 1024] f32
  const float* co     = (const float*)d_in[1];  // [512, 512] f32
  const float* counts = (const float*)d_in[2];  // [512] f32
  const float* emb    = (const float*)d_in[3];  // [512, 4] f32
  float* out = (float*)d_out;                   // [64, 512, 1024] f32

  unsigned short* ew = (unsigned short*)d_ws;   // 512*512 bf16 = 512 KB

  build_edge_weights<<<NCLS, 256, 0, stream>>>(co, counts, emb, ew);
  fused_graph_gate<<<NBATCH * (NCLS / 32), 512, 0, stream>>>(x, ew, out);
}
